// Attention_57767310131821
// MI455X (gfx1250) — matrix operand order repaired and checked by reading, not compiled, not run on hardware
//
#include <hip/hip_runtime.h>
#include <hip/hip_bf16.h>
#include <math.h>

typedef __attribute__((ext_vector_type(16))) _Float16 v16h;
typedef __attribute__((ext_vector_type(8)))  _Float16 v8h;
typedef __attribute__((ext_vector_type(8)))  float    v8f;

union V16 { v16h v; v8h h[2]; };

#define NTOK   4608      // b*t*h*w = 1*2*48*48
#define CDIM   256
#define HW     2304      // 48*48
#define WINTOK 2048      // t*32*32
#define QSCALE 0.17677669529663687f  // 1/sqrt(32)

// ---------------- weight fp32 -> f16 conversion ----------------
__global__ __launch_bounds__(256)
void cvt_w_kernel(const float* __restrict__ a, const float* __restrict__ b,
                  const float* __restrict__ c, const float* __restrict__ d,
                  _Float16* __restrict__ dst) {
    int idx = blockIdx.x * 256 + threadIdx.x;     // 4 * 65536
    const float* srcs[4] = {a, b, c, d};
    dst[idx] = (_Float16)srcs[idx >> 16][idx & 65535];
}

// ---------------- LayerNorm -> xn (f16) ----------------
__global__ __launch_bounds__(256)
void ln_kernel(const float* __restrict__ x, const float* __restrict__ ln_w,
               const float* __restrict__ ln_b, _Float16* __restrict__ xn) {
    int wave = threadIdx.x >> 5;
    int lane = threadIdx.x & 31;
    int token = blockIdx.x * 8 + wave;            // 576 blocks * 8 waves = 4608
    const float* p = x + (size_t)token * CDIM + lane * 8;
    float v[8];
    #pragma unroll
    for (int e = 0; e < 8; ++e) v[e] = p[e];
    float s = 0.f, ss = 0.f;
    #pragma unroll
    for (int e = 0; e < 8; ++e) { s += v[e]; ss += v[e] * v[e]; }
    #pragma unroll
    for (int o = 1; o < 32; o <<= 1) { s += __shfl_xor(s, o, 32); ss += __shfl_xor(ss, o, 32); }
    float mu  = s * (1.0f / 256.0f);
    float var = ss * (1.0f / 256.0f) - mu * mu;
    float rinv = rsqrtf(var + 1e-6f);
    _Float16* q = xn + (size_t)token * CDIM + lane * 8;
    #pragma unroll
    for (int e = 0; e < 8; ++e) {
        int ch = lane * 8 + e;
        q[e] = (_Float16)((v[e] - mu) * rinv * ln_w[ch] + ln_b[ch]);
    }
}

// ---------------- QKV GEMM: Y = xn @ W^T (WMMA f16->f32) ----------------
// which=0 -> Q (scaled, row-major f16), which=1 -> K (row-major f16),
// which=2 -> Vt (transposed: Vt[d][token], f16)
__global__ __launch_bounds__(32)
void qkv_kernel(const _Float16* __restrict__ xn, const _Float16* __restrict__ w16,
                _Float16* __restrict__ Qf, _Float16* __restrict__ Kf,
                _Float16* __restrict__ Vt) {
    int mt = blockIdx.x;                 // 288 tiles of M
    int nt = blockIdx.y;                 // 16 tiles of N
    int which = blockIdx.z;              // 3
    const _Float16* W = w16 + (size_t)which * 65536;
    int lane = threadIdx.x, colL = lane & 15, half = lane >> 4;
    v8f acc = {};
    #pragma unroll
    for (int kk = 0; kk < 8; ++kk) {
        int k0 = kk * 32;
        V16 a, b;
        const _Float16* ap = xn + (size_t)(mt * 16 + colL) * CDIM + k0;
        a.h[0] = *(const v8h*)(ap + 16 * half);
        a.h[1] = *(const v8h*)(ap + 16 * half + 8);
        // B[k][n] = W[n][k]: lane holds column n=colL, K chunk 16*half..+15 (contiguous in W row)
        const _Float16* bp = W + (size_t)(nt * 16 + colL) * CDIM + k0 + 16 * half;
        b.h[0] = *(const v8h*)bp;
        b.h[1] = *(const v8h*)(bp + 8);
        acc = __builtin_amdgcn_wmma_f32_16x16x32_f16(false, a.v, false, b.v, (short)0, acc, false, false);
    }
    #pragma unroll
    for (int r = 0; r < 8; ++r) {
        int row = mt * 16 + r + 8 * half;   // C layout: lane<16 rows 0..7, lane>=16 rows 8..15
        int col = nt * 16 + colL;
        float v = acc[r];
        if (which == 0)      Qf[(size_t)row * CDIM + col] = (_Float16)(v * QSCALE);
        else if (which == 1) Kf[(size_t)row * CDIM + col] = (_Float16)v;
        else                 Vt[(size_t)col * NTOK + row] = (_Float16)v;
    }
}

// ---------------- Flash attention per (window, head, 16-row q-tile) ----------------
__global__ __launch_bounds__(32)
void attn_kernel(const _Float16* __restrict__ Qf, const _Float16* __restrict__ Kf,
                 const _Float16* __restrict__ Vt, float* __restrict__ out_win) {
    __shared__ _Float16 lds[16 * 32];    // P-tile transpose staging (one wave per block)
    int bid = blockIdx.x;                // 4 windows * 8 heads * 128 qtiles = 4096
    int qt = bid & 127;
    int hh = (bid >> 7) & 7;
    int wn = bid >> 10;
    int wr = wn >> 1, wc = wn & 1;
    int rs = 16 * wr, cs = 16 * wc;      // window starts {0,16}
    int lane = threadIdx.x, colL = lane & 15, half = lane >> 4;

    // Q tile: 16 consecutive window tokens = same (t,i), contiguous j -> contiguous global tokens
    int sq = qt * 16;
    int tq = sq >> 10, remq = sq & 1023, iq = remq >> 5, jq = remq & 31;
    int gq = tq * HW + (rs + iq) * 48 + (cs + jq);
    V16 qa;
    {
        const _Float16* p = Qf + (size_t)(gq + colL) * CDIM + hh * 32;
        qa.h[0] = *(const v8h*)(p + 16 * half);
        qa.h[1] = *(const v8h*)(p + 16 * half + 8);
    }
    float m[8], l[8];
    v8f o0 = {}, o1 = {};
    #pragma unroll
    for (int r = 0; r < 8; ++r) { m[r] = -INFINITY; l[r] = 0.f; }

    for (int kc = 0; kc < 64; ++kc) {    // 64 chunks of 32 keys (one window row each)
        int sk = kc * 32;
        int tk = sk >> 10, ik = (sk & 1023) >> 5;
        int gk = tk * HW + (rs + ik) * 48 + cs;
        // K rows as B-operands for S = Q*K^T (lane holds K row gk+colL, d chunk by lane-half)
        V16 kb0, kb1;
        {
            const _Float16* p0 = Kf + (size_t)(gk + colL) * CDIM + hh * 32 + 16 * half;
            const _Float16* p1 = Kf + (size_t)(gk + 16 + colL) * CDIM + hh * 32 + 16 * half;
            kb0.h[0] = *(const v8h*)p0; kb0.h[1] = *(const v8h*)(p0 + 8);
            kb1.h[0] = *(const v8h*)p1; kb1.h[1] = *(const v8h*)(p1 + 8);
        }
        v8f zero = {};
        v8f s0 = __builtin_amdgcn_wmma_f32_16x16x32_f16(false, qa.v, false, kb0.v, (short)0, zero, false, false);
        v8f s1 = __builtin_amdgcn_wmma_f32_16x16x32_f16(false, qa.v, false, kb1.v, (short)0, zero, false, false);

        // online softmax: rows live in accumulator index, columns across each 16-lane half
        float e0[8], e1[8];
        #pragma unroll
        for (int r = 0; r < 8; ++r) {
            float v = fmaxf(s0[r], s1[r]);
            v = fmaxf(v, __shfl_xor(v, 1, 16));
            v = fmaxf(v, __shfl_xor(v, 2, 16));
            v = fmaxf(v, __shfl_xor(v, 4, 16));
            v = fmaxf(v, __shfl_xor(v, 8, 16));
            float mn = fmaxf(m[r], v);
            float alpha = __expf(m[r] - mn);
            m[r] = mn;
            e0[r] = __expf(s0[r] - mn);
            e1[r] = __expf(s1[r] - mn);
            float rsum = e0[r] + e1[r];
            rsum += __shfl_xor(rsum, 1, 16);
            rsum += __shfl_xor(rsum, 2, 16);
            rsum += __shfl_xor(rsum, 4, 16);
            rsum += __shfl_xor(rsum, 8, 16);
            l[r] = l[r] * alpha + rsum;
            o0[r] *= alpha; o1[r] *= alpha;
        }
        // P (C-layout) -> LDS row-major -> reload as A-operand
        #pragma unroll
        for (int r = 0; r < 8; ++r) {
            int row = r + 8 * half;
            lds[row * 32 + colL]      = (_Float16)e0[r];
            lds[row * 32 + 16 + colL] = (_Float16)e1[r];
        }
        asm volatile("s_wait_dscnt 0" ::: "memory");   // same-wave cross-lane LDS RAW
        V16 pa;
        {
            const _Float16* p = &lds[colL * 32];
            pa.h[0] = *(const v8h*)(p + 16 * half);
            pa.h[1] = *(const v8h*)(p + 16 * half + 8);
        }
        // V chunk as B-operands from transposed Vt (contiguous tokens per lane)
        V16 vb0, vb1;
        {
            const _Float16* p0 = Vt + (size_t)(hh * 32 + colL) * NTOK + gk + 16 * half;
            const _Float16* p1 = Vt + (size_t)(hh * 32 + 16 + colL) * NTOK + gk + 16 * half;
            vb0.h[0] = *(const v8h*)p0; vb0.h[1] = *(const v8h*)(p0 + 8);
            vb1.h[0] = *(const v8h*)p1; vb1.h[1] = *(const v8h*)(p1 + 8);
        }
        o0 = __builtin_amdgcn_wmma_f32_16x16x32_f16(false, pa.v, false, vb0.v, (short)0, o0, false, false);
        o1 = __builtin_amdgcn_wmma_f32_16x16x32_f16(false, pa.v, false, vb1.v, (short)0, o1, false, false);
        asm volatile("s_wait_dscnt 0" ::: "memory");   // drain before next-iter LDS overwrite
    }
    #pragma unroll
    for (int r = 0; r < 8; ++r) {
        int s = qt * 16 + r + 8 * half;
        float inv = 1.0f / l[r];
        size_t base = ((size_t)wn * WINTOK + s) * CDIM + hh * 32 + colL;
        out_win[base]      = o0[r] * inv;
        out_win[base + 16] = o1[r] * inv;
    }
}

// ---------------- overlap merge (gather, deterministic) -> merged f16 ----------------
__global__ __launch_bounds__(256)
void merge_kernel(const float* __restrict__ ow, _Float16* __restrict__ m16) {
    int idx = blockIdx.x * 256 + threadIdx.x;     // 4608 * 256
    int ch = idx & 255, token = idx >> 8;
    int t = token / HW, rc = token % HW;
    int r = rc / 48, c = rc % 48;
    float sum = 0.f; int cnt = 0;
    #pragma unroll
    for (int wr = 0; wr < 2; ++wr) {
        bool rok = (wr == 0) ? (r < 32) : (r >= 16);
        #pragma unroll
        for (int wc = 0; wc < 2; ++wc) {
            bool cok = (wc == 0) ? (c < 32) : (c >= 16);
            if (rok && cok) {
                int wn = wr * 2 + wc;
                int s = t * 1024 + (r - 16 * wr) * 32 + (c - 16 * wc);
                sum += ow[((size_t)wn * WINTOK + s) * CDIM + ch];
                cnt++;
            }
        }
    }
    m16[idx] = (_Float16)(sum / (float)cnt);
}

// ---------------- output GEMM: out = merged @ Wo^T + bo (WMMA) ----------------
__global__ __launch_bounds__(32)
void ogemm_kernel(const _Float16* __restrict__ m16, const _Float16* __restrict__ Wo,
                  const float* __restrict__ bo, float* __restrict__ out) {
    int mt = blockIdx.x, nt = blockIdx.y;
    int lane = threadIdx.x, colL = lane & 15, half = lane >> 4;
    float bias = bo[nt * 16 + colL];
    v8f acc;
    #pragma unroll
    for (int r = 0; r < 8; ++r) acc[r] = bias;
    #pragma unroll
    for (int kk = 0; kk < 8; ++kk) {
        int k0 = kk * 32;
        V16 a, b;
        const _Float16* ap = m16 + (size_t)(mt * 16 + colL) * CDIM + k0;
        a.h[0] = *(const v8h*)(ap + 16 * half);
        a.h[1] = *(const v8h*)(ap + 16 * half + 8);
        const _Float16* bp = Wo + (size_t)(nt * 16 + colL) * CDIM + k0 + 16 * half;
        b.h[0] = *(const v8h*)bp;
        b.h[1] = *(const v8h*)(bp + 8);
        acc = __builtin_amdgcn_wmma_f32_16x16x32_f16(false, a.v, false, b.v, (short)0, acc, false, false);
    }
    #pragma unroll
    for (int r = 0; r < 8; ++r)
        out[(size_t)(mt * 16 + r + 8 * half) * CDIM + nt * 16 + colL] = acc[r];
}

extern "C" void kernel_launch(void* const* d_in, const int* in_sizes, int n_in,
                              void* d_out, int out_size, void* d_ws, size_t ws_size,
                              hipStream_t stream) {
    const float* x    = (const float*)d_in[0];
    const float* ln_w = (const float*)d_in[1];
    const float* ln_b = (const float*)d_in[2];
    const float* Wq   = (const float*)d_in[3];
    const float* Wk   = (const float*)d_in[4];
    const float* Wv   = (const float*)d_in[5];
    const float* Wo   = (const float*)d_in[6];
    const float* bo   = (const float*)d_in[7];
    float* out = (float*)d_out;

    // workspace layout (bytes), total ~20.7 MB
    char* ws = (char*)d_ws;
    _Float16* xn  = (_Float16*)(ws + 0);         // 4608*256*2   = 2,359,296
    _Float16* w16 = (_Float16*)(ws + 2359296);   // 4*65536*2    =   524,288
    _Float16* Qf  = (_Float16*)(ws + 2883584);   // 2,359,296
    _Float16* Kf  = (_Float16*)(ws + 5242880);   // 2,359,296
    _Float16* Vt  = (_Float16*)(ws + 7602176);   // 2,359,296
    float*    ow  = (float*)   (ws + 9961472);   // 4*2048*256*4 = 8,388,608
    _Float16* m16 = (_Float16*)(ws + 18350080);  // 2,359,296

    cvt_w_kernel<<<1024, 256, 0, stream>>>(Wq, Wk, Wv, Wo, w16);
    ln_kernel<<<576, 256, 0, stream>>>(x, ln_w, ln_b, xn);
    qkv_kernel<<<dim3(288, 16, 3), 32, 0, stream>>>(xn, w16, Qf, Kf, Vt);
    attn_kernel<<<4096, 32, 0, stream>>>(Qf, Kf, Vt, ow);
    merge_kernel<<<4608, 256, 0, stream>>>(ow, m16);
    ogemm_kernel<<<dim3(288, 16), 32, 0, stream>>>(m16, w16 + 3 * 65536, bo, out);
}
